// MultiHeadAttention_12635793785029
// MI455X (gfx1250) — matrix operand order repaired and checked by reading, not compiled, not run on hardware
//
#include <hip/hip_runtime.h>
#include <hip/hip_bf16.h>
#include <stdint.h>

typedef __attribute__((ext_vector_type(16))) __bf16 v16bf;
typedef __attribute__((ext_vector_type(8)))  float  v8f;

#define BATCH 8
#define CH    512
#define TLEN  1024
#define NH    8
#define KCH   64

struct __align__(16) U4 { unsigned int x, y, z, w; };

union Frag16 {
  v16bf v;
  U4    q[2];
};

__device__ __forceinline__ unsigned short f2bf(float f) {
  unsigned int u = __builtin_bit_cast(unsigned int, f);
  unsigned int r = u + 0x7FFFu + ((u >> 16) & 1u);
  return (unsigned short)(r >> 16);
}

// ---------------- Kernel A: weights fp32 -> bf16, packed [4][512][512] ----------------
__global__ void k_convert_w(const float* __restrict__ wq, const float* __restrict__ wk,
                            const float* __restrict__ wv, const float* __restrict__ wo,
                            unsigned short* __restrict__ out) {
  int idx = blockIdx.x * blockDim.x + threadIdx.x;     // 0 .. 4*2^18
  if (idx >= 4 * CH * CH) return;
  int mat = idx >> 18;
  int off = idx & (CH * CH - 1);
  const float* src = (mat == 0) ? wq : (mat == 1) ? wk : (mat == 2) ? wv : wo;
  out[idx] = f2bf(src[off]);
}

// ------- Kernel B: x/c fp32 [b][ch][t] -> bf16 transposed [b][t][ch] (LDS tiles) -------
__global__ void k_transpose(const float* __restrict__ xin, const float* __restrict__ cin,
                            unsigned short* __restrict__ xT, unsigned short* __restrict__ cT) {
  __shared__ float tile[32][33];
  int bid   = blockIdx.x;
  int which = bid >> 12;              // 0: x, 1: c (4096 tiles each)
  int rem   = bid & 4095;
  int b     = rem >> 9;
  int r2    = rem & 511;
  int cht   = r2 >> 5;                // 16 channel tiles of 32
  int ttl   = r2 & 31;                // 32 t tiles of 32
  const float* src = (which ? cin : xin) + (size_t)b * CH * TLEN + (size_t)(cht * 32) * TLEN + ttl * 32;
  unsigned short* dst = (which ? cT : xT) + (size_t)b * TLEN * CH + (size_t)(ttl * 32) * CH + cht * 32;
  int tc = threadIdx.x & 31;
  int tr = threadIdx.x >> 5;          // 0..7
#pragma unroll
  for (int p = 0; p < 4; ++p) {
    int row = tr + p * 8;
    tile[row][tc] = src[(size_t)row * TLEN + tc];
  }
  __syncthreads();
#pragma unroll
  for (int p = 0; p < 4; ++p) {
    int trow = tr + p * 8;
    dst[(size_t)trow * CH + tc] = f2bf(tile[tc][trow]);
  }
}

// ------- Kernel C: QKV projection (bf16 WMMA) + bias + fused RoPE -------
// Each wave: one (b, mat, head, t-tile16) -> 64ch x 16t tile = 4 f32 accumulators.
// Software-pipelined K-loop; sched_barrier pins prefetch loads above the WMMAs so
// the waitcnt inserter can use partial s_wait_loadcnt values.
__global__ void __launch_bounds__(128)
k_qkv(const unsigned short* __restrict__ Wb,
      const unsigned short* __restrict__ xT, const unsigned short* __restrict__ cT,
      const float* __restrict__ bq, const float* __restrict__ bk, const float* __restrict__ bv,
      unsigned short* __restrict__ Qb, unsigned short* __restrict__ Kb,
      unsigned short* __restrict__ Vt) {
  __shared__ __align__(16) unsigned char sQK[4][2048];
  int lane = threadIdx.x & 31;
  int wid  = threadIdx.x >> 5;
  int unit = blockIdx.x * 4 + wid;    // ((b*3+mat)*8+h)*64+tt
  int tt   = unit & 63;
  int h    = (unit >> 6) & 7;
  int m3   = unit >> 9;
  int mat  = m3 % 3;
  int b    = m3 / 3;
  int hi   = lane >> 4;
  int ln   = lane & 15;

  const unsigned short* W   = Wb + (size_t)mat * CH * CH;
  const unsigned short* src = ((mat == 0) ? xT : cT) + (size_t)b * TLEN * CH;
  const float* bias = (mat == 0) ? bq : (mat == 1) ? bk : bv;

  v8f acc[4] = {};
  int trow = tt * 16 + ln;
  const unsigned char* srow = (const unsigned char*)(src + (size_t)trow * CH);

  auto loadB = [&](int c0, Frag16& f) {
    const unsigned char* bp = srow + c0 * 2 + (hi ? 16 : 0);
    f.q[0] = *(const U4*)(bp);
    f.q[1] = *(const U4*)(bp + 32);
  };
  auto loadA = [&](int c0, Frag16* f) {
#pragma unroll
    for (int a = 0; a < 4; ++a) {
      int orow = h * 64 + a * 16 + ln;
      const unsigned char* ap = (const unsigned char*)(W + (size_t)orow * CH + c0) + hi * 16;
      f[a].q[0] = *(const U4*)(ap);
      f[a].q[1] = *(const U4*)(ap + 32);
    }
  };

  Frag16 bcur, bnxt, acur[4], anxt[4];
  loadB(0, bcur);
  loadA(0, acur);
#pragma unroll
  for (int c0 = 0; c0 < CH; c0 += 32) {
    if (c0 + 32 < CH) { loadB(c0 + 32, bnxt); loadA(c0 + 32, anxt); }
    __builtin_amdgcn_sched_barrier(0);   // keep c+1 loads issued before chunk-c WMMAs
#pragma unroll
    for (int a = 0; a < 4; ++a)
      acc[a] = __builtin_amdgcn_wmma_f32_16x16x32_bf16(false, acur[a].v, false, bcur.v,
                                                       (short)0, acc[a], false, false);
    if (c0 + 32 < CH) {
      bcur = bnxt;
#pragma unroll
      for (int a = 0; a < 4; ++a) acur[a] = anxt[a];
    }
  }
  // bias (C layout: vgpr r -> channel a*16 + r + 8*hi, lane -> t)
#pragma unroll
  for (int a = 0; a < 4; ++a)
#pragma unroll
    for (int r = 0; r < 8; ++r)
      acc[a][r] += bias[h * 64 + a * 16 + r + 8 * hi];

  if (mat < 2) {
    // RoPE: pair kc (acc0) with kc+16 (acc1), same vgpr slot. d=32, 16 thetas.
#pragma unroll
    for (int r = 0; r < 8; ++r) {
      int   j     = r + 8 * hi;                        // kc in [0,16)
      float theta = __expf(-0.5756462732485114f * (float)j);  // 10000^(-2j/32)
      float ang   = (float)trow * theta;
      float sn, cs;
      __sincosf(ang, &sn, &cs);
      float a0 = acc[0][r], a1 = acc[1][r];
      acc[0][r] = a0 * cs - a1 * sn;
      acc[1][r] = a1 * cs + a0 * sn;
    }
    // Assemble [t][kc] tile in LDS, then fully-coalesced 2KB store.
    unsigned char* tb = sQK[wid];
#pragma unroll
    for (int a = 0; a < 4; ++a)
#pragma unroll
      for (int r = 0; r < 8; ++r) {
        int kc = a * 16 + r + 8 * hi;
        *(unsigned short*)(tb + ln * 128 + kc * 2) = f2bf(acc[a][r]);
      }
    __asm__ volatile("s_wait_dscnt 0" ::: "memory");
    unsigned short* dq = ((mat == 0) ? Qb : Kb) + ((size_t)(b * NH + h) * TLEN + tt * 16) * KCH;
    unsigned char* go = (unsigned char*)dq;
#pragma unroll
    for (int j = 0; j < 4; ++j) {
      U4 d = *(const U4*)(tb + (lane + j * 32) * 16);
      *(U4*)(go + (lane + j * 32) * 16) = d;
    }
  } else {
    // V transposed [kc][t]: per vgpr 16 consecutive t.
    unsigned short* dv = Vt + (size_t)(b * NH + h) * KCH * TLEN;
#pragma unroll
    for (int a = 0; a < 4; ++a)
#pragma unroll
      for (int r = 0; r < 8; ++r) {
        int kc = a * 16 + r + 8 * hi;
        dv[(size_t)kc * TLEN + tt * 16 + ln] = f2bf(acc[a][r]);
      }
  }
}

// ------- Kernel D: flash attention per wave (16 q-rows x full K_CH=64) -------
__global__ void __launch_bounds__(128)
k_attn(const unsigned short* __restrict__ Qb, const unsigned short* __restrict__ Kb,
       const unsigned short* __restrict__ Vt, unsigned short* __restrict__ att) {
  __shared__ __align__(16) unsigned char sP[4][1024];  // per-wave 16x32 bf16 P tile
  int lane = threadIdx.x & 31;
  int wid  = threadIdx.x >> 5;
  int unit = blockIdx.x * 4 + wid;    // ((b*8)+h)*64+tt
  int tt   = unit & 63;
  int h    = (unit >> 6) & 7;
  int b    = unit >> 9;
  int hi   = lane >> 4;
  int ln   = lane & 15;
  int bh   = b * NH + h;

  // Q tile as two A fragments (kc 0-31, 32-63), persistent in registers.
  const unsigned char* qrow = (const unsigned char*)(Qb + ((size_t)bh * TLEN + tt * 16 + ln) * KCH);
  Frag16 A0, A1;
  A0.q[0] = *(const U4*)(qrow + hi * 16);
  A0.q[1] = *(const U4*)(qrow + 32 + hi * 16);
  A1.q[0] = *(const U4*)(qrow + 64 + hi * 16);
  A1.q[1] = *(const U4*)(qrow + 96 + hi * 16);

  v8f O[4] = {};
  float mrow[8], lrow[8], corr[8];
#pragma unroll
  for (int r = 0; r < 8; ++r) { mrow[r] = -1e30f; lrow[r] = 0.f; }

  unsigned char* pt = sP[wid];
  const unsigned char* kbase = (const unsigned char*)(Kb + (size_t)bh * TLEN * KCH);
  const unsigned char* vbase = (const unsigned char*)(Vt + (size_t)bh * KCH * TLEN);

  // K fragments for one 32-key block: [tile0 kc0-31, tile0 kc32-63, tile1 kc0-31, tile1 kc32-63]
  auto loadK = [&](int s0, Frag16* f) {
    const unsigned char* k0 = kbase + (size_t)(s0 + ln) * 128 + hi * 16;
    f[0].q[0] = *(const U4*)(k0);       f[0].q[1] = *(const U4*)(k0 + 32);
    f[1].q[0] = *(const U4*)(k0 + 64);  f[1].q[1] = *(const U4*)(k0 + 96);
    const unsigned char* k1 = kbase + (size_t)(s0 + 16 + ln) * 128 + hi * 16;
    f[2].q[0] = *(const U4*)(k1);       f[2].q[1] = *(const U4*)(k1 + 32);
    f[3].q[0] = *(const U4*)(k1 + 64);  f[3].q[1] = *(const U4*)(k1 + 96);
  };

  Frag16 Kc[4], Kn[4];
  loadK(0, Kc);

#pragma unroll 2
  for (int s0 = 0; s0 < TLEN; s0 += 32) {
    v8f S0 = {}, S1 = {};
    S0 = __builtin_amdgcn_wmma_f32_16x16x32_bf16(false, A0.v, false, Kc[0].v, (short)0, S0, false, false);
    S0 = __builtin_amdgcn_wmma_f32_16x16x32_bf16(false, A1.v, false, Kc[1].v, (short)0, S0, false, false);
    S1 = __builtin_amdgcn_wmma_f32_16x16x32_bf16(false, A0.v, false, Kc[2].v, (short)0, S1, false, false);
    S1 = __builtin_amdgcn_wmma_f32_16x16x32_bf16(false, A1.v, false, Kc[3].v, (short)0, S1, false, false);

    // V fragments for this block (consumed after softmax), then next key block.
    // Both load groups overlap the softmax VALU work below.
    Frag16 Bv[4];
#pragma unroll
    for (int a = 0; a < 4; ++a) {                      // B = V^T (k=s 32, n=kc 16)
      const unsigned char* vr = vbase + ((size_t)(a * 16 + ln) * TLEN + s0 + hi * 16) * 2;
      Bv[a].q[0] = *(const U4*)(vr);
      Bv[a].q[1] = *(const U4*)(vr + 16);
    }
    if (s0 + 32 < TLEN) loadK(s0 + 32, Kn);
    __builtin_amdgcn_sched_barrier(0);   // keep prefetches issued before the softmax

    // Online softmax: row stats live per vgpr slot r (row = r + 8*hi).
#pragma unroll
    for (int r = 0; r < 8; ++r) {
      float a0 = S0[r] * 0.125f;        // 1/sqrt(64)
      float a1 = S1[r] * 0.125f;
      float mx = fmaxf(a0, a1);
      mx = fmaxf(mx, __shfl_xor(mx, 1, 32));
      mx = fmaxf(mx, __shfl_xor(mx, 2, 32));
      mx = fmaxf(mx, __shfl_xor(mx, 4, 32));
      mx = fmaxf(mx, __shfl_xor(mx, 8, 32));
      float mn = fmaxf(mrow[r], mx);
      float cf = __expf(mrow[r] - mn);
      float e0 = __expf(a0 - mn);
      float e1 = __expf(a1 - mn);
      int   m  = r + 8 * hi;            // write P tile [m][k] to LDS (C->A relayout)
      *(unsigned short*)(pt + m * 64 + ln * 2)      = f2bf(e0);
      *(unsigned short*)(pt + m * 64 + 32 + ln * 2) = f2bf(e1);
      float s = e0 + e1;
      s += __shfl_xor(s, 1, 32);
      s += __shfl_xor(s, 2, 32);
      s += __shfl_xor(s, 4, 32);
      s += __shfl_xor(s, 8, 32);
      lrow[r] = lrow[r] * cf + s;
      mrow[r] = mn;
      corr[r] = cf;
    }
#pragma unroll
    for (int a = 0; a < 4; ++a)
#pragma unroll
      for (int r = 0; r < 8; ++r) O[a][r] *= corr[r];

    __asm__ volatile("s_wait_dscnt 0" ::: "memory");   // same-wave LDS RAW
    Frag16 Ap;                                         // P as A fragment (16x32)
    {
      const unsigned char* pr = pt + ln * 64 + hi * 32;
      Ap.q[0] = *(const U4*)(pr);
      Ap.q[1] = *(const U4*)(pr + 16);
    }
#pragma unroll
    for (int a = 0; a < 4; ++a)
      O[a] = __builtin_amdgcn_wmma_f32_16x16x32_bf16(false, Ap.v, false, Bv[a].v, (short)0, O[a], false, false);
    __asm__ volatile("" ::: "memory");                 // keep reads before next-iter writes
    if (s0 + 32 < TLEN) {
#pragma unroll
      for (int i = 0; i < 4; ++i) Kc[i] = Kn[i];
    }
  }
  // Normalize and store att[b][t][c] bf16 (c = h*64 + kc).
  unsigned short* ab = att + (size_t)b * TLEN * CH;
#pragma unroll
  for (int r = 0; r < 8; ++r) lrow[r] = 1.0f / lrow[r];
#pragma unroll
  for (int a = 0; a < 4; ++a)
#pragma unroll
    for (int r = 0; r < 8; ++r) {
      int t = tt * 16 + r + 8 * hi;
      ab[(size_t)t * CH + h * 64 + a * 16 + ln] = f2bf(O[a][r] * lrow[r]);
    }
}

// ------- Kernel E: output projection (bf16 WMMA) + bias, fp32 out [b][o][t] -------
__global__ void __launch_bounds__(128)
k_oproj(const unsigned short* __restrict__ Wb, const unsigned short* __restrict__ att,
        const float* __restrict__ bo, float* __restrict__ out) {
  int lane = threadIdx.x & 31;
  int wid  = threadIdx.x >> 5;
  int unit = blockIdx.x * 4 + wid;    // ((b*8)+oc)*64+tt
  int tt   = unit & 63;
  int oc   = (unit >> 6) & 7;
  int b    = unit >> 9;
  int hi   = lane >> 4;
  int ln   = lane & 15;

  const unsigned short* W = Wb + (size_t)3 * CH * CH;
  const unsigned char* arow = (const unsigned char*)(att + ((size_t)b * TLEN + tt * 16 + ln) * CH);

  auto loadB = [&](int c0, Frag16& f) {
    const unsigned char* bp = arow + c0 * 2 + (hi ? 16 : 0);
    f.q[0] = *(const U4*)(bp);
    f.q[1] = *(const U4*)(bp + 32);
  };
  auto loadA = [&](int c0, Frag16* f) {
#pragma unroll
    for (int a = 0; a < 4; ++a) {
      int orow = oc * 64 + a * 16 + ln;
      const unsigned char* ap = (const unsigned char*)(W + (size_t)orow * CH + c0) + hi * 16;
      f[a].q[0] = *(const U4*)(ap);
      f[a].q[1] = *(const U4*)(ap + 32);
    }
  };

  v8f acc[4] = {};
  Frag16 bcur, bnxt, acur[4], anxt[4];
  loadB(0, bcur);
  loadA(0, acur);
#pragma unroll
  for (int c0 = 0; c0 < CH; c0 += 32) {
    if (c0 + 32 < CH) { loadB(c0 + 32, bnxt); loadA(c0 + 32, anxt); }
    __builtin_amdgcn_sched_barrier(0);   // keep c+1 loads issued before chunk-c WMMAs
#pragma unroll
    for (int a = 0; a < 4; ++a)
      acc[a] = __builtin_amdgcn_wmma_f32_16x16x32_bf16(false, acur[a].v, false, bcur.v,
                                                       (short)0, acc[a], false, false);
    if (c0 + 32 < CH) {
      bcur = bnxt;
#pragma unroll
      for (int a = 0; a < 4; ++a) acur[a] = anxt[a];
    }
  }
#pragma unroll
  for (int a = 0; a < 4; ++a)
#pragma unroll
    for (int r = 0; r < 8; ++r) {
      int o = oc * 64 + a * 16 + r + 8 * hi;
      out[((size_t)b * CH + o) * TLEN + tt * 16 + ln] = acc[a][r] + bo[o];
    }
}

extern "C" void kernel_launch(void* const* d_in, const int* in_sizes, int n_in,
                              void* d_out, int out_size, void* d_ws, size_t ws_size,
                              hipStream_t stream) {
  const float* x  = (const float*)d_in[0];
  const float* c  = (const float*)d_in[1];
  const float* Wq = (const float*)d_in[2];
  const float* bq = (const float*)d_in[3];
  const float* Wk = (const float*)d_in[4];
  const float* bk = (const float*)d_in[5];
  const float* Wv = (const float*)d_in[6];
  const float* bv = (const float*)d_in[7];
  const float* Wo = (const float*)d_in[8];
  const float* bo = (const float*)d_in[9];
  float* out = (float*)d_out;

  unsigned char* ws = (unsigned char*)d_ws;
  size_t off = 0;
  unsigned short* Wb  = (unsigned short*)(ws + off); off += (size_t)4 * CH * CH * 2;        // 2 MB
  unsigned short* xT  = (unsigned short*)(ws + off); off += (size_t)BATCH * TLEN * CH * 2;  // 8 MB
  unsigned short* cT  = (unsigned short*)(ws + off); off += (size_t)BATCH * TLEN * CH * 2;  // 8 MB
  unsigned short* Qb  = (unsigned short*)(ws + off); off += (size_t)BATCH * NH * TLEN * KCH * 2;
  unsigned short* Kb  = (unsigned short*)(ws + off); off += (size_t)BATCH * NH * TLEN * KCH * 2;
  unsigned short* Vt  = (unsigned short*)(ws + off); off += (size_t)BATCH * NH * KCH * TLEN * 2;
  unsigned short* att = (unsigned short*)(ws + off); off += (size_t)BATCH * TLEN * CH * 2;
  (void)ws_size; (void)in_sizes; (void)n_in; (void)out_size;

  k_convert_w<<<(4 * CH * CH) / 256, 256, 0, stream>>>(Wq, Wk, Wv, Wo, Wb);
  k_transpose<<<8192, 256, 0, stream>>>(x, c, xT, cT);
  k_qkv<<<3072, 128, 0, stream>>>(Wb, xT, cT, bq, bk, bv, Qb, Kb, Vt);
  k_attn<<<1024, 128, 0, stream>>>(Qb, Kb, Vt, att);
  k_oproj<<<1024, 128, 0, stream>>>(Wb, att, bo, out);
}
